// DeepSeekMoELayer_54391465837035
// MI455X (gfx1250) — matrix operand order repaired and checked by reading, not compiled, not run on hardware
//
#include <hip/hip_runtime.h>
#include <math.h>

#define DEVINL __device__ __forceinline__

typedef __bf16 v16bf __attribute__((ext_vector_type(16)));
typedef __bf16 v8bf  __attribute__((ext_vector_type(8)));
typedef float  v8f   __attribute__((ext_vector_type(8)));

constexpr int Bdim = 4, Sdim = 2048, Hdim = 1024, Idim = 256, Edim = 256, TOPK = 2, CAP = 256;
constexpr int T    = Bdim * Sdim;   // 8192 tokens
constexpr int NASN = T * TOPK;      // 16384 assignments

constexpr int AS   = 48;   // LDS stride (elems) for A tiles  (96B, 16B-aligned rows)
constexpr int BS   = 48;   // LDS stride (elems) for B tiles
constexpr int ACTS = 272;  // LDS stride (elems) for act tile (544B rows)

// ---------------- WMMA fragment helpers (wave32, gfx1250 layouts) ----------------

// A (16x32 bf16, MxK): lane = M (0..15) in each half; lanes 0-15 hold K[0..7],K[16..23],
// lanes 16-31 hold K[8..15],K[24..31]  -> two contiguous 8-elem (16B) chunks per lane.
DEVINL v16bf lds_a_frag(const __bf16* base, int stride, int lane) {
  const int m = lane & 15, khi = lane >> 4;
  const __bf16* p = base + m * stride + khi * 8;
  union { v16bf v; v8bf h[2]; } u;
  u.h[0] = *(const v8bf*)p;
  u.h[1] = *(const v8bf*)(p + 16);
  return u.v;
}

// B (32x16 bf16, KxN) stored transposed in LDS as [N][K]: lane n = N, lanes 0-15 hold
// K[0..7],K[16..23], lanes 16-31 hold K[8..15],K[24..31] -> two 8-elem chunks per lane.
DEVINL v16bf lds_b_frag(const __bf16* base, int stride, int lane) {
  const int n = lane & 15, kh = lane >> 4;
  const __bf16* p = base + n * stride + kh * 8;
  union { v16bf v; v8bf h[2]; } u;
  u.h[0] = *(const v8bf*)p;
  u.h[1] = *(const v8bf*)(p + 16);
  return u.v;
}

DEVINL v8f wmma_bf16(v16bf a, v16bf b, v8f c) {
  // (neg_a, A, neg_b, B, c_mod, C, reuse_a, reuse_b)
  return __builtin_amdgcn_wmma_f32_16x16x32_bf16(false, a, false, b, (short)0, c, false, false);
}

// ---------------- kernel 0: zero expert counters ----------------
__global__ void k_zero_cnt(int* __restrict__ cnt) { cnt[threadIdx.x] = 0; }

// ---------------- kernel 1: router (softmax + top-2 + slot assignment) ----------------
__global__ void __launch_bounds__(256) k_router(const float* __restrict__ x,
                                                const float* __restrict__ rw,
                                                int* __restrict__ cnt,
                                                int* __restrict__ asn,
                                                float* __restrict__ wts) {
  __shared__ alignas(16) float xs[Hdim];
  __shared__ float fred[Edim];
  __shared__ unsigned long long kred[Edim];

  const int t = blockIdx.x;
  const int e = threadIdx.x;

#pragma unroll
  for (int j = 0; j < Hdim / Edim; ++j)
    xs[j * Edim + e] = x[(size_t)t * Hdim + j * Edim + e];
  __syncthreads();

  const float4* xs4 = (const float4*)xs;
  const float4* rw4 = (const float4*)(rw + (size_t)e * Hdim);
  float acc = 0.f;
#pragma unroll 8
  for (int h = 0; h < Hdim / 4; ++h) {
    float4 a = xs4[h], b = rw4[h];
    acc += a.x * b.x + a.y * b.y + a.z * b.z + a.w * b.w;
  }

  // full softmax over experts
  fred[e] = acc; __syncthreads();
  for (int s = Edim / 2; s > 0; s >>= 1) {
    if (e < s) fred[e] = fmaxf(fred[e], fred[e + s]);
    __syncthreads();
  }
  const float mx = fred[0]; __syncthreads();
  float p = __expf(acc - mx);
  fred[e] = p; __syncthreads();
  for (int s = Edim / 2; s > 0; s >>= 1) {
    if (e < s) fred[e] += fred[e + s];
    __syncthreads();
  }
  const float inv = 1.f / fred[0]; __syncthreads();
  p *= inv;  // prob in (0,1] -> positive, uint bit pattern is order-preserving

  unsigned long long key =
      ((unsigned long long)__float_as_uint(p) << 32) | (unsigned)(Edim - 1 - e);
  kred[e] = key; __syncthreads();
  for (int s = Edim / 2; s > 0; s >>= 1) {
    if (e < s) kred[e] = kred[e] > kred[e + s] ? kred[e] : kred[e + s];
    __syncthreads();
  }
  const unsigned long long k1 = kred[0]; __syncthreads();
  const int e1 = (Edim - 1) - (int)(k1 & 0xFFu);
  kred[e] = (e == e1) ? 0ull : key; __syncthreads();
  for (int s = Edim / 2; s > 0; s >>= 1) {
    if (e < s) kred[e] = kred[e] > kred[e + s] ? kred[e] : kred[e + s];
    __syncthreads();
  }
  const unsigned long long k2 = kred[0];

  if (e == 0) {
    const int e2 = (Edim - 1) - (int)(k2 & 0xFFu);
    const float p1 = __uint_as_float((unsigned)(k1 >> 32));
    const float p2 = __uint_as_float((unsigned)(k2 >> 32));
    const float m = fmaxf(p1, p2);
    float w1 = __expf(p1 - m), w2 = __expf(p2 - m);
    const float s = w1 + w2; w1 /= s; w2 /= s;
    const int a1 = t * 2, a2 = t * 2 + 1;
    int s1 = atomicAdd(&cnt[e1], 1); s1 = min(s1, CAP - 1);
    asn[e1 * CAP + s1] = a1; wts[a1] = w1;
    int s2 = atomicAdd(&cnt[e2], 1); s2 = min(s2, CAP - 1);
    asn[e2 * CAP + s2] = a2; wts[a2] = w2;
  }
}

// ---------------- kernel 2: fused per-expert SwiGLU FFN (bf16 WMMA) ----------------
// grid: (E, CAP/64), block: 512 threads = 16 waves.
// wave w: msub = w&3 (16-row M-subtile), ngrp = w>>2 (64-col N-group).
__global__ void __launch_bounds__(512) k_ffn(const float* __restrict__ x,
                                             const float* __restrict__ wg,
                                             const float* __restrict__ wu,
                                             const float* __restrict__ wd,
                                             const float* __restrict__ gsc,
                                             const float* __restrict__ usc,
                                             const float* __restrict__ dsc,
                                             const int* __restrict__ cnt,
                                             const int* __restrict__ asn,
                                             const float* __restrict__ wts,
                                             float* __restrict__ yasn) {
  __shared__ alignas(16) __bf16 sA[64 * AS];         //  6.0 KB
  __shared__ alignas(16) __bf16 sBg[Idim * BS];      // 24.0 KB
  __shared__ alignas(16) __bf16 sBu[Idim * BS];      // 24.0 KB
  __shared__ alignas(16) __bf16 sAct[64 * ACTS];     // 34.0 KB
  __shared__ int sAsn[64];
  __shared__ int sTok[64];
  __shared__ float sWt[64];

  const int e  = blockIdx.x;
  const int m0 = blockIdx.y * 64;
  const int cntE = min(cnt[e], CAP);
  if (m0 >= cntE) return;
  const int rows = min(64, cntE - m0);

  const int tid  = threadIdx.x;
  const int lane = tid & 31;
  const int wave = tid >> 5;
  const int msub = wave & 3;
  const int ngrp = wave >> 2;

  if (tid < 64) {
    const int s = m0 + tid;
    if (s < cntE) {
      const int a = asn[e * CAP + s];
      sAsn[tid] = a; sTok[tid] = a >> 1; sWt[tid] = wts[a];
    } else {
      sAsn[tid] = 0; sTok[tid] = 0; sWt[tid] = 0.f;  // padded rows: valid addr, never written out
    }
  }
  __syncthreads();

  v8f accG[4] = {};
  v8f accU[4] = {};

  // ---- phase 1: gate/up GEMMs, K over H ----
  for (int k0 = 0; k0 < Hdim; k0 += 32) {
    // stage A: 64x32 fp32 -> bf16, one float4 per thread
    {
      const int r = tid >> 3, c4 = (tid & 7) * 4;
      const float4 v = *(const float4*)(x + (size_t)sTok[r] * Hdim + k0 + c4);
      sA[r * AS + c4 + 0] = (__bf16)v.x;
      sA[r * AS + c4 + 1] = (__bf16)v.y;
      sA[r * AS + c4 + 2] = (__bf16)v.z;
      sA[r * AS + c4 + 3] = (__bf16)v.w;
    }
    // stage Bg/Bu: 32x256 fp32 -> bf16, transposed to [i][k]
    const size_t baseg = ((size_t)e * Hdim + k0) * Idim;
#pragma unroll
    for (int j = 0; j < 4; ++j) {
      const int idx = j * 512 + tid;          // 2048 float4s
      const int kk = idx >> 6, i4 = (idx & 63) * 4;
      const float4 g = *(const float4*)(wg + baseg + (size_t)kk * Idim + i4);
      const float4 u = *(const float4*)(wu + baseg + (size_t)kk * Idim + i4);
      sBg[(i4 + 0) * BS + kk] = (__bf16)g.x;
      sBg[(i4 + 1) * BS + kk] = (__bf16)g.y;
      sBg[(i4 + 2) * BS + kk] = (__bf16)g.z;
      sBg[(i4 + 3) * BS + kk] = (__bf16)g.w;
      sBu[(i4 + 0) * BS + kk] = (__bf16)u.x;
      sBu[(i4 + 1) * BS + kk] = (__bf16)u.y;
      sBu[(i4 + 2) * BS + kk] = (__bf16)u.z;
      sBu[(i4 + 3) * BS + kk] = (__bf16)u.w;
    }
    __syncthreads();

    const v16bf aF = lds_a_frag(sA + msub * 16 * AS, AS, lane);
#pragma unroll
    for (int ns = 0; ns < 4; ++ns) {
      const int n = ngrp * 64 + ns * 16;
      const v16bf bg = lds_b_frag(sBg + n * BS, BS, lane);
      accG[ns] = wmma_bf16(aF, bg, accG[ns]);
      const v16bf bu = lds_b_frag(sBu + n * BS, BS, lane);
      accU[ns] = wmma_bf16(aF, bu, accU[ns]);
    }
    __syncthreads();
  }

  // ---- activation: act = silu(gate*gs) * (up*us), stored bf16 in LDS ----
  {
    const float g_s = gsc[e], u_s = usc[e];
    const int mbase = msub * 16 + ((lane >> 4) << 3);  // C layout: M = r + 8*(lane>=16)
    const int nbase = ngrp * 64 + (lane & 15);
#pragma unroll
    for (int ns = 0; ns < 4; ++ns) {
#pragma unroll
      for (int r = 0; r < 8; ++r) {
        const float g = accG[ns][r] * g_s;
        const float u = accU[ns][r] * u_s;
        const float a = (g / (1.f + __expf(-g))) * u;
        sAct[(mbase + r) * ACTS + nbase + ns * 16] = (__bf16)a;
      }
    }
  }
  __syncthreads();

  // ---- phase 2: down projection, N over H in 4 chunks of 256 ----
  const float d_s = dsc[e];
  for (int nchunk = 0; nchunk < 4; ++nchunk) {
    v8f accD[4] = {};
    for (int k0 = 0; k0 < Idim; k0 += 32) {
      const size_t based = ((size_t)e * Idim + k0) * Hdim + nchunk * 256;
#pragma unroll
      for (int j = 0; j < 4; ++j) {
        const int idx = j * 512 + tid;
        const int kk = idx >> 6, i4 = (idx & 63) * 4;
        const float4 d = *(const float4*)(wd + based + (size_t)kk * Hdim + i4);
        sBg[(i4 + 0) * BS + kk] = (__bf16)d.x;
        sBg[(i4 + 1) * BS + kk] = (__bf16)d.y;
        sBg[(i4 + 2) * BS + kk] = (__bf16)d.z;
        sBg[(i4 + 3) * BS + kk] = (__bf16)d.w;
      }
      __syncthreads();

      const v16bf aF = lds_a_frag(sAct + msub * 16 * ACTS + k0, ACTS, lane);
#pragma unroll
      for (int ns = 0; ns < 4; ++ns) {
        const v16bf bd = lds_b_frag(sBg + (ngrp * 64 + ns * 16) * BS, BS, lane);
        accD[ns] = wmma_bf16(aF, bd, accD[ns]);
      }
      __syncthreads();
    }
    // write y_asn[a] = y * down_scale * routing_weight (deterministic; unique rows)
    const int mbase = msub * 16 + ((lane >> 4) << 3);
    const int nbase = nchunk * 256 + ngrp * 64 + (lane & 15);
#pragma unroll
    for (int ns = 0; ns < 4; ++ns) {
#pragma unroll
      for (int r = 0; r < 8; ++r) {
        const int m = mbase + r;
        if (m < rows) {
          const float v = accD[ns][r] * d_s * sWt[m];
          yasn[(size_t)sAsn[m] * Hdim + nbase + ns * 16] = v;
        }
      }
    }
  }
}

// ---------------- kernel 3: combine top-2 assignments ----------------
__global__ void __launch_bounds__(256) k_combine(const float* __restrict__ yasn,
                                                 float* __restrict__ out) {
  const int tid = blockIdx.x * 256 + threadIdx.x;     // over T*H/4 float4s
  const int t = tid >> 8;                             // H/4 = 256 vec4 per token
  const int c = tid & 255;
  const float4* y4 = (const float4*)yasn;
  float4* o4 = (float4*)out;
  const float4 a = y4[(size_t)(2 * t) * 256 + c];
  const float4 b = y4[(size_t)(2 * t + 1) * 256 + c];
  o4[tid] = make_float4(a.x + b.x, a.y + b.y, a.z + b.z, a.w + b.w);
}

// ---------------- host entry ----------------
extern "C" void kernel_launch(void* const* d_in, const int* in_sizes, int n_in,
                              void* d_out, int out_size, void* d_ws, size_t ws_size,
                              hipStream_t stream) {
  (void)in_sizes; (void)n_in; (void)out_size; (void)ws_size;
  const float* x      = (const float*)d_in[0];  // [B,S,H]
  const float* rw     = (const float*)d_in[1];  // [E,H]
  const float* wg     = (const float*)d_in[2];  // [E,H,I]
  const float* wu     = (const float*)d_in[3];  // [E,H,I]
  const float* wd     = (const float*)d_in[4];  // [E,I,H]
  const float* gscale = (const float*)d_in[5];  // [E]
  const float* uscale = (const float*)d_in[6];  // [E]
  const float* dscale = (const float*)d_in[7];  // [E]
  float* out = (float*)d_out;

  // workspace layout
  char* ws = (char*)d_ws;
  int*   cnt  = (int*)ws;                                               // E ints
  int*   asn  = (int*)(ws + 1024);                                      // E*CAP ints
  float* wts  = (float*)(ws + 1024 + (size_t)Edim * CAP * 4);           // N floats
  float* yasn = (float*)(ws + 1024 + (size_t)Edim * CAP * 4 + (size_t)NASN * 4);  // N*H floats

  hipLaunchKernelGGL(k_zero_cnt, dim3(1), dim3(Edim), 0, stream, cnt);
  hipLaunchKernelGGL(k_router, dim3(T), dim3(Edim), 0, stream, x, rw, cnt, asn, wts);
  hipLaunchKernelGGL(k_ffn, dim3(Edim, CAP / 64), dim3(512), 0, stream,
                     x, wg, wu, wd, gscale, uscale, dscale, cnt, asn, wts, yasn);
  hipLaunchKernelGGL(k_combine, dim3((T * Hdim / 4) / 256), dim3(256), 0, stream, yasn, out);
}
